// MultiheadAttention_84250078478921
// MI455X (gfx1250) — compile-verified, matrix operand order checked by reading, not run on hardware
//
#include <hip/hip_runtime.h>
#include <hip/hip_bf16.h>

// ---------------------------------------------------------------------------
// MHA forward for MI455X (gfx1250): bf16 WMMA GEMMs + flash attention.
// B=2, N=2048, D=1024, H=16, HD=64.
// Round 2: async global->LDS double buffering, DPP softmax reductions,
//          softmax scale folded into Q.
// ---------------------------------------------------------------------------

typedef __attribute__((ext_vector_type(16))) __bf16 v16bf;
typedef __attribute__((ext_vector_type(8)))  __bf16 v8bf;
typedef __attribute__((ext_vector_type(8)))  float  v8f;

#define MHA_B  2
#define MHA_N  2048
#define MHA_D  1024
#define MHA_H  16
#define MHA_HD 64
#define MHA_M  (MHA_B * MHA_N)          // 4096 rows
#define MHA_SCALE 0.125f                // 64^-0.5 (folded into Q)

#if defined(__gfx1250__)
#define MHA_ASYNC_LDS 1
#else
#define MHA_ASYNC_LDS 0
#endif

static __device__ inline v8f vzero8f() {
    v8f z = {0.f, 0.f, 0.f, 0.f, 0.f, 0.f, 0.f, 0.f};
    return z;
}

// ---------------------------------------------------------------------------
// Async global->LDS 16B copy (CDNA5 GLOBAL_LOAD_ASYNC_TO_LDS_B128, ASYNCcnt).
// Per-lane: copies 16 bytes from the global address to the LDS address. Fallback: sync VGPR round-trip.
// ---------------------------------------------------------------------------
static __device__ inline void async_copy_b128(const __bf16* g, __bf16* lds) {
#if MHA_ASYNC_LDS
    asm volatile("global_load_async_to_lds_b128 %0, %1, off"
                 :
                 : "v"((unsigned)(unsigned long long)lds), "v"(g)
                 : "memory");
#else
    *(uint4*)lds = *(const uint4*)g;
#endif
}

static __device__ inline void async_wait0() {
#if MHA_ASYNC_LDS
    asm volatile("s_wait_asynccnt 0x0" ::: "memory");
#endif
}

// ---------------------------------------------------------------------------
// DPP16 row reductions: full reduce across a 16-lane row (lanes 0-15 and
// 16-31 reduce independently) with row_ror 8/4/2/1. Pure VALU -> co-executes
// with WMMA (XDL ops are TRANS-class), no LDS-unit pressure.
// ---------------------------------------------------------------------------
template <int CTRL>
static __device__ inline float dpp_rot(float x) {
    return __int_as_float(__builtin_amdgcn_update_dpp(
        0, __float_as_int(x), CTRL, 0xF, 0xF, true));
}
static __device__ inline float row16_max(float v) {
    v = fmaxf(v, dpp_rot<0x128>(v));  // row_ror:8
    v = fmaxf(v, dpp_rot<0x124>(v));  // row_ror:4
    v = fmaxf(v, dpp_rot<0x122>(v));  // row_ror:2
    v = fmaxf(v, dpp_rot<0x121>(v));  // row_ror:1
    return v;
}
static __device__ inline float row16_sum(float v) {
    v += dpp_rot<0x128>(v);
    v += dpp_rot<0x124>(v);
    v += dpp_rot<0x122>(v);
    v += dpp_rot<0x121>(v);
    return v;
}

// Fragment loader: 16x32 bf16 A-layout (ISA 7.12.2).
// Lane l holds row (l&15); lane half (l>>4) selects K-chunks
// [hi*8, hi*8+8) and [16+hi*8, 16+hi*8+8). Two 16B loads -> 8 VGPRs.
// Works for A fragments and (via B^T storage) for B fragments.
static __device__ inline v16bf load_frag(const __bf16* tile, int stride, int lane) {
    int r  = lane & 15;
    int hi = (lane >> 4) & 1;
    v8bf c0 = *(const v8bf*)(tile + (size_t)r * stride + hi * 8);
    v8bf c1 = *(const v8bf*)(tile + (size_t)r * stride + 16 + hi * 8);
    return __builtin_shufflevector(c0, c1, 0, 1, 2, 3, 4, 5, 6, 7,
                                   8, 9, 10, 11, 12, 13, 14, 15);
}

static __device__ inline v8f wmma_bf16(v16bf a, v16bf b, v8f c) {
    return __builtin_amdgcn_wmma_f32_16x16x32_bf16(
        /*neg_a=*/false, a, /*neg_b=*/false, b,
        /*c_mod=*/(short)0, c, /*reuse_a=*/false, /*reuse_b=*/false);
}

// ---------------------------------------------------------------------------
// Kernel: f32 -> bf16 elementwise convert (n multiple of 4)
// ---------------------------------------------------------------------------
__global__ void convert_bf16_kernel(const float* __restrict__ in,
                                    __bf16* __restrict__ out, int n) {
    int i = (blockIdx.x * blockDim.x + threadIdx.x) * 4;
    if (i < n) {
        float4 v = *(const float4*)(in + i);
        out[i + 0] = (__bf16)v.x;
        out[i + 1] = (__bf16)v.y;
        out[i + 2] = (__bf16)v.z;
        out[i + 3] = (__bf16)v.w;
    }
}

// ---------------------------------------------------------------------------
// Kernel: f32 [rows][cols] -> bf16 transposed [cols][rows] (B^T layout)
// ---------------------------------------------------------------------------
__global__ void transpose_bf16_kernel(const float* __restrict__ W,
                                      __bf16* __restrict__ WT,
                                      int rows, int cols) {
    int i = blockIdx.x * blockDim.x + threadIdx.x;
    if (i < rows * cols) {
        int r = i / cols;
        int c = i - r * cols;
        WT[(size_t)c * rows + r] = (__bf16)W[i];
    }
}

// ---------------------------------------------------------------------------
// Tiled bf16 WMMA GEMM: C[M][Nc] = A[M][K] * B[K][Nc] + bias, B given as B^T.
// 128x128 tile / block, K-step 32, 8 waves, each wave: 2x4 16x16 C tiles.
// Double-buffered async global->LDS staging overlaps DMA with WMMA.
// MODE 0: f32 output.  MODE 1: QKV epilogue (scatter Q*SCALE, K [B,H,N,HD]
//                               bf16 and V transposed [B,H,HD,N] bf16).
// ---------------------------------------------------------------------------
template <int MODE>
__global__ __launch_bounds__(256) void gemm_bf16_kernel(
    const __bf16* __restrict__ A,   // [M][K]
    const __bf16* __restrict__ BT,  // [Nc][K]
    const float* __restrict__ bias, // [Nc]
    float* __restrict__ Cout,       // MODE 0
    __bf16* __restrict__ Qb,        // MODE 1
    __bf16* __restrict__ Kb,        // MODE 1
    __bf16* __restrict__ Vt,        // MODE 1
    int M, int Nc, int K) {
    __shared__ __bf16 As[2][128 * 32];
    __shared__ __bf16 Bs[2][128 * 32];

    const int t    = threadIdx.x;
    const int wave = t >> 5;
    const int lane = t & 31;
    const int wm   = wave & 3;   // 0..3 -> 32-row slab
    const int wn   = wave >> 2;  // 0..1 -> 64-col slab
    const int blockN = blockIdx.x * 128;
    const int blockM = blockIdx.y * 128;

    v8f acc[2][4];
#pragma unroll
    for (int i = 0; i < 2; ++i)
#pragma unroll
        for (int j = 0; j < 4; ++j) acc[i][j] = vzero8f();

    // Cooperative staging map: each thread DMAs 32B of A and 32B of B.
    const int srow  = t >> 1;
    const int selem = (t & 1) * 16;

    auto stage = [&](int k0, int p) {
        const __bf16* ag = A + (size_t)(blockM + srow) * K + k0 + selem;
        __bf16* la = &As[p][srow * 32 + selem];
        async_copy_b128(ag, la);
        async_copy_b128(ag + 8, la + 8);
        const __bf16* bg = BT + (size_t)(blockN + srow) * K + k0 + selem;
        __bf16* lb = &Bs[p][srow * 32 + selem];
        async_copy_b128(bg, lb);
        async_copy_b128(bg + 8, lb + 8);
    };

    stage(0, 0);
    async_wait0();
    __syncthreads();

    int p = 0;
    for (int k0 = 0; k0 < K; k0 += 32) {
        if (k0 + 32 < K) stage(k0 + 32, p ^ 1);  // prefetch overlaps WMMA

        v16bf af[2], bf[4];
#pragma unroll
        for (int i = 0; i < 2; ++i)
            af[i] = load_frag(&As[p][(wm * 32 + i * 16) * 32], 32, lane);
#pragma unroll
        for (int j = 0; j < 4; ++j)
            bf[j] = load_frag(&Bs[p][(wn * 64 + j * 16) * 32], 32, lane);
#pragma unroll
        for (int i = 0; i < 2; ++i)
#pragma unroll
            for (int j = 0; j < 4; ++j)
                acc[i][j] = wmma_bf16(af[i], bf[j], acc[i][j]);

        async_wait0();
        __syncthreads();
        p ^= 1;
    }

    // Epilogue. C fragment: lane l, element r -> row (l>>4)*8 + r, col l&15.
    const int lr = lane & 15;
    const int lh = lane >> 4;
#pragma unroll
    for (int i = 0; i < 2; ++i) {
#pragma unroll
        for (int j = 0; j < 4; ++j) {
            int rbase = blockM + wm * 32 + i * 16 + lh * 8;
            int col   = blockN + wn * 64 + j * 16 + lr;
            float bv  = bias[col];
#pragma unroll
            for (int r = 0; r < 8; ++r) {
                float v = acc[i][j][r] + bv;
                int row = rbase + r;
                if (MODE == 0) {
                    Cout[(size_t)row * Nc + col] = v;
                } else {
                    // col in [0,3072): which/head/hd split; row = b*N + n
                    int which = col >> 10;
                    int h     = (col >> 6) & (MHA_H - 1);
                    int hd    = col & (MHA_HD - 1);
                    int b     = row >> 11;
                    int n     = row & (MHA_N - 1);
                    size_t bh = (size_t)(b * MHA_H + h);
                    if (which == 0)  // fold softmax scale into Q
                        Qb[(bh * MHA_N + n) * MHA_HD + hd] =
                            (__bf16)(v * MHA_SCALE);
                    else if (which == 1)
                        Kb[(bh * MHA_N + n) * MHA_HD + hd] = (__bf16)v;
                    else
                        Vt[(bh * MHA_HD + hd) * MHA_N + n] = (__bf16)v;
                }
            }
        }
    }
}

// ---------------------------------------------------------------------------
// Flash attention: grid (N/128, B*H), 8 waves. Wave owns 16 query rows,
// Q fragments resident; loop over 32-key blocks, double-buffered async LDS
// staging. Per block: 4 WMMA scores + DPP online softmax + 4 WMMA PV.
// Output context written bf16 as [B][N][D] (row-major A for out-proj GEMM).
// ---------------------------------------------------------------------------
__global__ __launch_bounds__(256) void flash_attn_kernel(
    const __bf16* __restrict__ Qb,  // [B*H][N][HD] (pre-scaled by HD^-0.5)
    const __bf16* __restrict__ Kb,  // [B*H][N][HD]
    const __bf16* __restrict__ Vt,  // [B*H][HD][N]
    __bf16* __restrict__ Ctx) {     // [B][N][D]
    __shared__ __bf16 Ks[2][32 * 64];    // [key][hd]
    __shared__ __bf16 Vs[2][64 * 32];    // [hd][key]
    __shared__ __bf16 Ps[8][16 * 32];    // per-wave P staging

    const int t    = threadIdx.x;
    const int wave = t >> 5;
    const int lane = t & 31;
    const int lr   = lane & 15;
    const int lh   = lane >> 4;
    const int bh   = blockIdx.y;                   // b*H + h
    const int qbase = blockIdx.x * 128 + wave * 16;

    const __bf16* Qh = Qb + (size_t)bh * MHA_N * MHA_HD;
    const __bf16* Kh = Kb + (size_t)bh * MHA_N * MHA_HD;
    const __bf16* Vh = Vt + (size_t)bh * MHA_HD * MHA_N;

    // Resident Q fragments (16 rows x 64 = 2 K-steps of 32)
    v16bf qf[2];
#pragma unroll
    for (int ks = 0; ks < 2; ++ks)
        qf[ks] = load_frag(Qh + (size_t)qbase * MHA_HD + ks * 32, MHA_HD, lane);

    v8f oacc[4];
#pragma unroll
    for (int c = 0; c < 4; ++c) oacc[c] = vzero8f();
    float mrun[8], lrun[8];
#pragma unroll
    for (int r = 0; r < 8; ++r) { mrun[r] = -1e30f; lrun[r] = 0.f; }

    // Cooperative staging: K block 32x64 (4KB) + V^T block 64x32 (4KB).
    auto stage = [&](int kb, int p) {
        {
            int r = t >> 3, c = (t & 7) * 8;
            async_copy_b128(Kh + (size_t)(kb + r) * MHA_HD + c,
                            &Ks[p][r * 64 + c]);
        }
        {
            int r = t >> 2, c = (t & 3) * 8;
            async_copy_b128(Vh + (size_t)r * MHA_N + kb + c,
                            &Vs[p][r * 32 + c]);
        }
    };

    stage(0, 0);
    async_wait0();
    __syncthreads();

    int p = 0;
    for (int kb = 0; kb < MHA_N; kb += 32) {
        if (kb + 32 < MHA_N) stage(kb + 32, p ^ 1);  // prefetch next block

        // Scores: S[16 x 32] = Q(16x64) * K(32 keys x 64)^T (scale pre-folded)
        v8f s[2];
        s[0] = vzero8f();
        s[1] = vzero8f();
#pragma unroll
        for (int j = 0; j < 2; ++j)
#pragma unroll
            for (int ks = 0; ks < 2; ++ks)
                s[j] = wmma_bf16(
                    qf[ks],
                    load_frag(&Ks[p][(j * 16) * 64 + ks * 32], 64, lane),
                    s[j]);

        // Online softmax; row m of the C fragment lives on the 16-lane row
        // sharing (lane>>4) -> DPP row_ror reductions, no LDS traffic.
        float alpha[8];
#pragma unroll
        for (int r = 0; r < 8; ++r) {
            float v0 = s[0][r];
            float v1 = s[1][r];
            float mx = row16_max(fmaxf(v0, v1));
            float nm = fmaxf(mrun[r], mx);
            alpha[r] = __expf(mrun[r] - nm);
            float p0 = __expf(v0 - nm);
            float p1 = __expf(v1 - nm);
            s[0][r] = p0;
            s[1][r] = p1;
            lrun[r] = lrun[r] * alpha[r] + row16_sum(p0 + p1);
            mrun[r] = nm;
        }
#pragma unroll
        for (int c = 0; c < 4; ++c)
#pragma unroll
            for (int r = 0; r < 8; ++r) oacc[c][r] *= alpha[r];

        // Re-layout P (C fragment -> A fragment) through per-wave LDS.
        __bf16* pw = &Ps[wave][0];
#pragma unroll
        for (int j = 0; j < 2; ++j)
#pragma unroll
            for (int r = 0; r < 8; ++r)
                pw[(lh * 8 + r) * 32 + j * 16 + lr] = (__bf16)s[j][r];
        // Same-wave LDS ops are serviced in order (DScnt); no barrier needed.
        v16bf pf = load_frag(pw, 32, lane);

        // PV: O(16x64) += P(16x32) * V(32x64); B fragment from V^T rows.
#pragma unroll
        for (int c = 0; c < 4; ++c)
            oacc[c] = wmma_bf16(pf, load_frag(&Vs[p][(c * 16) * 32], 32, lane),
                                oacc[c]);

        async_wait0();
        __syncthreads();
        p ^= 1;
    }

    // Normalize and write context bf16 to [B][N][D].
    const int b = bh >> 4;
    const int h = bh & (MHA_H - 1);
    float inv[8];
#pragma unroll
    for (int r = 0; r < 8; ++r) inv[r] = (lrun[r] > 0.f) ? (1.0f / lrun[r]) : 0.f;
#pragma unroll
    for (int c = 0; c < 4; ++c)
#pragma unroll
        for (int r = 0; r < 8; ++r) {
            int n   = qbase + lh * 8 + r;
            int col = h * MHA_HD + c * 16 + lr;
            Ctx[((size_t)(b * MHA_N + n)) * MHA_D + col] =
                (__bf16)(oacc[c][r] * inv[r]);
        }
}

// ---------------------------------------------------------------------------
// Host-side launch
// ---------------------------------------------------------------------------
extern "C" void kernel_launch(void* const* d_in, const int* in_sizes, int n_in,
                              void* d_out, int out_size, void* d_ws,
                              size_t ws_size, hipStream_t stream) {
    (void)in_sizes; (void)n_in; (void)out_size; (void)ws_size;
    const float* query = (const float*)d_in[0];  // [B,N,D]
    const float* Wqkv  = (const float*)d_in[1];  // [D,3D]
    const float* bqkv  = (const float*)d_in[2];  // [3D]
    const float* Wo    = (const float*)d_in[3];  // [D,D]
    const float* bo    = (const float*)d_in[4];  // [D]
    float* out         = (float*)d_out;          // [B,N,D]

    char* ws = (char*)d_ws;
    size_t off = 0;
    __bf16* Abf   = (__bf16*)(ws + off); off += (size_t)MHA_M * MHA_D * 2;     // 8 MB
    __bf16* WqkvT = (__bf16*)(ws + off); off += (size_t)3 * MHA_D * MHA_D * 2; // 6 MB
    __bf16* WoT   = (__bf16*)(ws + off); off += (size_t)MHA_D * MHA_D * 2;     // 2 MB
    __bf16* Qb    = (__bf16*)(ws + off); off += (size_t)MHA_M * MHA_D * 2;     // 8 MB
    __bf16* Kbuf  = (__bf16*)(ws + off); off += (size_t)MHA_M * MHA_D * 2;     // 8 MB
    __bf16* Vt    = (__bf16*)(ws + off); off += (size_t)MHA_M * MHA_D * 2;     // 8 MB
    __bf16* Ctx   = (__bf16*)(ws + off); off += (size_t)MHA_M * MHA_D * 2;     // 8 MB

    // 1) Precision conversion / weight transposition.
    {
        int n = MHA_M * MHA_D;  // 4194304
        convert_bf16_kernel<<<n / (256 * 4), 256, 0, stream>>>(query, Abf, n);
    }
    {
        int n = MHA_D * 3 * MHA_D;  // 3145728
        transpose_bf16_kernel<<<(n + 255) / 256, 256, 0, stream>>>(
            Wqkv, WqkvT, MHA_D, 3 * MHA_D);
    }
    {
        int n = MHA_D * MHA_D;  // 1048576
        transpose_bf16_kernel<<<(n + 255) / 256, 256, 0, stream>>>(
            Wo, WoT, MHA_D, MHA_D);
    }

    // 2) QKV projection GEMM (+bias) with head-split / V-transpose epilogue.
    gemm_bf16_kernel<1><<<dim3((3 * MHA_D) / 128, MHA_M / 128), 256, 0, stream>>>(
        Abf, WqkvT, bqkv, nullptr, Qb, Kbuf, Vt, MHA_M, 3 * MHA_D, MHA_D);

    // 3) Flash attention.
    flash_attn_kernel<<<dim3(MHA_N / 128, MHA_B * MHA_H), 256, 0, stream>>>(
        Qb, Kbuf, Vt, Ctx);

    // 4) Output projection GEMM (+bias), f32 output.
    gemm_bf16_kernel<0><<<dim3(MHA_D / 128, MHA_M / 128), 256, 0, stream>>>(
        Ctx, WoT, bo, out, nullptr, nullptr, nullptr, MHA_M, MHA_D, MHA_D);
}
